// SAGCN_model_83150566851145
// MI455X (gfx1250) — matrix operand order repaired and checked by reading, not compiled, not run on hardware
//
#include <hip/hip_runtime.h>
#include <hip/hip_bf16.h>
#include <math.h>

#define BS_   1024
#define NP_   256     // NUM_PATCH (L)
#define PP_   128     // PATCH
#define HID_  256
#define ATT_  128
#define IND_  40

typedef __attribute__((ext_vector_type(16))) __bf16 v16bf;
typedef __attribute__((ext_vector_type(8)))  float  v8f;
typedef __attribute__((ext_vector_type(4)))  int    v4i;
typedef __attribute__((address_space(1))) v4i as1_v4i;
typedef __attribute__((address_space(3))) v4i as3_v4i;

union FragU { v16bf v; unsigned int u[8]; };

// CDNA5 async-copy path (GLOBAL_LOAD_ASYNC_TO_LDS_B128, ASYNCcnt-tracked)
#if defined(__gfx1250__)
#if __has_builtin(__builtin_amdgcn_global_load_async_to_lds_b128)
#define ASYNC_LDS 1
#endif
#endif

__device__ __forceinline__ void wait_async_all() {
#ifdef ASYNC_LDS
#if __has_builtin(__builtin_amdgcn_s_wait_asynccnt)
    __builtin_amdgcn_s_wait_asynccnt(0);
#else
    asm volatile("s_wait_asynccnt 0" ::: "memory");
#endif
#endif
}

// ---------------- block reductions (shared-mem tree, all threads get result) --------
__device__ __forceinline__ float redSum(float* sh, int n, float v) {
    int t = threadIdx.x;
    sh[t] = v; __syncthreads();
    for (int s = n >> 1; s > 0; s >>= 1) {
        if (t < s) sh[t] = sh[t] + sh[t + s];
        __syncthreads();
    }
    float r = sh[0]; __syncthreads();
    return r;
}
__device__ __forceinline__ float redMax(float* sh, int n, float v) {
    int t = threadIdx.x;
    sh[t] = v; __syncthreads();
    for (int s = n >> 1; s > 0; s >>= 1) {
        if (t < s) sh[t] = fmaxf(sh[t], sh[t + s]);
        __syncthreads();
    }
    float r = sh[0]; __syncthreads();
    return r;
}
__device__ __forceinline__ float redMin(float* sh, int n, float v) {
    int t = threadIdx.x;
    sh[t] = v; __syncthreads();
    for (int s = n >> 1; s > 0; s >>= 1) {
        if (t < s) sh[t] = fminf(sh[t], sh[t + s]);
        __syncthreads();
    }
    float r = sh[0]; __syncthreads();
    return r;
}

// ---------------- 1) per-row feature extraction: 12 temporal + 8 spectral ----------
__global__ __launch_bounds__(128) void features_kernel(const float* __restrict__ x,
                                                       float* __restrict__ featRaw)
{
    __shared__ float sh[128];
    __shared__ float sg[128];
    __shared__ float re[128], im[128], ps[128];
    __shared__ float rv[128];
    __shared__ int   ri[128];
    __shared__ float shMed;

    const int t = threadIdx.x;
    const long long row = blockIdx.x;          // b*NP_ + l
    float s = x[row * PP_ + t];
    sg[t] = s;
    __syncthreads();

    // temporal statistics
    float mx   = redMax(sh, 128, s);
    float mn   = redMin(sh, 128, s);
    float sum  = redSum(sh, 128, s);
    float mean = sum * (1.f / 128.f);
    float sum2 = redSum(sh, 128, s * s);
    float rms  = sqrtf(sum2 * (1.f / 128.f));
    float cen  = s - mean;
    float var1 = redSum(sh, 128, cen * cen) * (1.f / 127.f);
    float sd   = sqrtf(var1);
    float ptp  = mx - mn;
    // softmax entropy
    float e  = expf(s - mx);
    float Z  = redSum(sh, 128, e);
    float T  = redSum(sh, 128, (s - mx) * e);
    float ent = logf(Z) - T / Z;
    // arcsin / arctan std (ddof=1)
    float uc = fminf(fmaxf(s, -1.f + 1e-7f), 1.f - 1e-7f);
    float ua = asinf(uc);
    float mu = redSum(sh, 128, ua) * (1.f / 128.f);
    float sis = sqrtf(redSum(sh, 128, (ua - mu) * (ua - mu)) * (1.f / 127.f));
    float ub = atanf(s);
    float mb = redSum(sh, 128, ub) * (1.f / 128.f);
    float sit = sqrtf(redSum(sh, 128, (ub - mb) * (ub - mb)) * (1.f / 127.f));
    float m4 = redSum(sh, 128, cen * cen * cen * cen) * (1.f / 128.f);
    float m3 = redSum(sh, 128, cen * cen * cen) * (1.f / 128.f);
    float kur = m4 / (var1 * var1) - 3.f;
    float skw = m3 / (sd * sd * sd);

    // 128-pt radix-2 DIT FFT in LDS
    int br = (int)(__brev((unsigned)t) >> 25);
    re[t] = sg[br];
    im[t] = 0.f;
    __syncthreads();
    for (int len = 2; len <= 128; len <<= 1) {
        int hl = len >> 1;
        if (t < 64) {
            int grp = t / hl;
            int pos = t - grp * hl;
            int i = grp * len + pos, j = i + hl;
            float ang = -6.28318530717958647692f * (float)pos / (float)len;
            float wr = cosf(ang), wi = sinf(ang);
            float tr = re[j] * wr - im[j] * wi;
            float ti = re[j] * wi + im[j] * wr;
            re[j] = re[i] - tr; im[j] = im[i] - ti;
            re[i] = re[i] + tr; im[i] = im[i] + ti;
        }
        __syncthreads();
    }
    float pr = re[t], pi = im[t];
    float a2  = pr * pr + pi * pi;
    float amp = sqrtf(a2);
    float psd = a2 * (1.f / 128.f);
    ps[t] = psd;
    __syncthreads();
    float fr = (t < 64) ? (float)t * (1.f / 128.f) : (float)(t - 128) * (1.f / 128.f);

    float psum  = redSum(sh, 128, psd);
    float meanf = redSum(sh, 128, fr * psd) / psum;
    float occ   = redSum(sh, 128, (fr < 0.5f) ? psd : 0.f) / psum;
    float pbw   = sqrtf(redSum(sh, 128, psd * psd) / psum);
    float mpsd  = redMax(sh, 128, psd);

    // argmax(amp) with first-index tie-break
    rv[t] = amp; ri[t] = t; __syncthreads();
    for (int sft = 64; sft > 0; sft >>= 1) {
        if (t < sft) {
            float ov = rv[t + sft]; int oi = ri[t + sft];
            if (ov > rv[t] || (ov == rv[t] && oi < ri[t])) { rv[t] = ov; ri[t] = oi; }
        }
        __syncthreads();
    }
    float mamp = rv[0];
    int   ia   = ri[0];
    float fma  = (ia < 64) ? (float)ia * (1.f / 128.f) : (float)(ia - 128) * (1.f / 128.f);
    __syncthreads();

    // median frequency: stable rank-64 selection on psd
    float mine = ps[t];
    int rank = 0;
    for (int j = 0; j < 128; ++j) {
        float o = ps[j];
        rank += (o < mine) || (o == mine && j < t);
    }
    if (rank == 64) shMed = fr;
    __syncthreads();
    float medf = shMed;

    if (t == 0) {
        float* o = featRaw + row * 20;
        o[0]=mx; o[1]=mn; o[2]=sd; o[3]=rms; o[4]=mean; o[5]=ptp; o[6]=var1; o[7]=ent;
        o[8]=sis; o[9]=sit; o[10]=kur; o[11]=skw;
        o[12]=meanf; o[13]=medf; o[14]=psum; o[15]=occ; o[16]=pbw; o[17]=mpsd; o[18]=mamp; o[19]=fma;
    }
}

// ---------------- 2) cumsum along L + signed-sqrt compress, assemble 40-d ----------
__global__ void cumsum_kernel(const float* __restrict__ raw, float* __restrict__ f40)
{
    int idx = blockIdx.x * blockDim.x + threadIdx.x;   // b*20 + d
    if (idx >= BS_ * 20) return;
    int b = idx / 20, d = idx - b * 20;
    float c = 0.f;
    for (int l = 0; l < NP_; ++l) {
        long long o = (long long)b * NP_ + l;
        float v = raw[o * 20 + d];
        c += v;
        float cu = c / sqrtf(fmaxf(fabsf(c), 1e-12f));
        f40[o * 40 + d]      = v;
        f40[o * 40 + 20 + d] = cu;
    }
}

// ---------------- 3) per-batch Frobenius normalization ------------------------------
__global__ __launch_bounds__(256) void norm_kernel(float* __restrict__ f40)
{
    __shared__ float sh[256];
    __shared__ float s_inv;
    int b = blockIdx.x, t = threadIdx.x;
    long long base = (long long)b * NP_ * 40;
    float acc = 0.f;
    for (int i = t; i < NP_ * 40; i += 256) { float v = f40[base + i]; acc += v * v; }
    float ss = redSum(sh, 256, acc);
    if (t == 0) s_inv = 1.f / sqrtf(ss);
    __syncthreads();
    float inv = s_inv;
    for (int i = t; i < NP_ * 40; i += 256) f40[base + i] *= inv;
}

// ---------------- pack: fp32 -> bf16 (optionally transposed), batched ---------------
// dst[b][r][c] = src[b][ trans ? c*ldsrc + r : r*ldsrc + c ]
__global__ __launch_bounds__(256) void pack_bf16_kernel(
    const float* __restrict__ src, long long sS, int ldsrc,
    __bf16* __restrict__ dst, long long sD,
    int R, int C, int trans)
{
    long long total = (long long)R * C;
    long long i = (long long)blockIdx.x * blockDim.x + threadIdx.x;
    if (i >= total) return;
    int b = blockIdx.y;
    int r = (int)(i / C), c = (int)(i - (long long)r * C);
    const float* s = src + (long long)b * sS;
    float v = trans ? s[(long long)c * ldsrc + r] : s[(long long)r * ldsrc + c];
    dst[(long long)b * sD + i] = (__bf16)v;
}

// ---------------- 4) batched GEMM: C = act(A * B^T + bias) --------------------------
// A: [M][K] bf16 (lda), B: [N][K] bf16 (ldb)  -- both K-major, pre-packed.
// fp32 WMMA accumulation via v_wmma_f32_16x16x32_bf16; LDS double-buffered with
// GLOBAL_LOAD_ASYNC_TO_LDS_B128 on the full-tile path.
#define BLKM 128
#define BLKN 128
#define KC   64
#define LDSA 80                       // row stride (elements): 160B, 16B-aligned rows
#define BUFSZ ((BLKM + BLKN) * LDSA)  // one double-buffer slot (A tile + B tile)

__device__ __forceinline__ void stage_scalar(const __bf16* __restrict__ Ab, int lda,
                                             const __bf16* __restrict__ Bb, int ldb,
                                             int m0, int n0, int k0,
                                             int M, int N, int K,
                                             __bf16* As, __bf16* Bs, int tid)
{
    for (int e = tid; e < BLKM * KC; e += 256) {
        int m = e >> 6, k = e & (KC - 1);
        int gm = m0 + m, gk = k0 + k;
        __bf16 v = (__bf16)0.f;
        if (gm < M && gk < K) v = Ab[(long long)gm * lda + gk];
        As[m * LDSA + k] = v;
    }
    for (int e = tid; e < BLKN * KC; e += 256) {
        int n = e >> 6, k = e & (KC - 1);
        int gn = n0 + n, gk = k0 + k;
        __bf16 v = (__bf16)0.f;
        if (gn < N && gk < K) v = Bb[(long long)gn * ldb + gk];
        Bs[n * LDSA + k] = v;
    }
}

__device__ __forceinline__ void stage_async(const __bf16* __restrict__ Ab, int lda,
                                            const __bf16* __restrict__ Bb, int ldb,
                                            int m0, int n0, int k0,
                                            __bf16* As, __bf16* Bs, int tid)
{
#ifdef ASYNC_LDS
    #pragma unroll
    for (int i = 0; i < (BLKM * KC / 8) / 256; ++i) {     // 4 iters, 16B per lane
        int e = tid + i * 256;
        int row = e >> 3, cb = e & 7;
        __builtin_amdgcn_global_load_async_to_lds_b128(
            (as1_v4i*)(Ab + (long long)(m0 + row) * lda + k0 + cb * 8),
            (as3_v4i*)(As + row * LDSA + cb * 8), 0, 0);
    }
    #pragma unroll
    for (int i = 0; i < (BLKN * KC / 8) / 256; ++i) {
        int e = tid + i * 256;
        int row = e >> 3, cb = e & 7;
        __builtin_amdgcn_global_load_async_to_lds_b128(
            (as1_v4i*)(Bb + (long long)(n0 + row) * ldb + k0 + cb * 8),
            (as3_v4i*)(Bs + row * LDSA + cb * 8), 0, 0);
    }
#endif
}

// one K-chunk of math: A-frag + all 8 B-frags loaded first, then 8 WMMAs back-to-back
__device__ __forceinline__ void tile_compute(const __bf16* As, const __bf16* Bs,
                                             v8f (&acc)[8], int wave, int half, int l16)
{
    const int mrow = wave * 16 + l16;
    #pragma unroll
    for (int ks = 0; ks < 2; ++ks) {
        const int kb = ks * 32;
        FragU fa;                                  // 16-bit A 16x32 layout (ISA 7.12.2)
        #pragma unroll
        for (int p = 0; p < 8; ++p) {
            int kk = (p < 4) ? (half * 8 + 2 * p) : (16 + half * 8 + 2 * (p - 4));
            fa.u[p] = *(const unsigned int*)&As[mrow * LDSA + kb + kk];
        }
        FragU fb[8];                               // 16-bit B 32x16 layout
        #pragma unroll
        for (int nt = 0; nt < 8; ++nt) {
            const int nrow = nt * 16 + l16;
            #pragma unroll
            for (int p = 0; p < 8; ++p)
                fb[nt].u[p] = *(const unsigned int*)&Bs[nrow * LDSA + kb + ((p < 4) ? (half * 8 + 2 * p) : (16 + half * 8 + 2 * (p - 4)))];
        }
        #pragma unroll
        for (int nt = 0; nt < 8; ++nt)
            acc[nt] = __builtin_amdgcn_wmma_f32_16x16x32_bf16(
                false, fa.v, false, fb[nt].v, (short)0, acc[nt], false, false);
    }
}

__global__ __launch_bounds__(256) void gemm_bf16_kernel(
    const __bf16* __restrict__ A, long long sA, int lda,
    const __bf16* __restrict__ B, long long sB, int ldb,
    const float* __restrict__ bias, int biasMode,     // 0 none, 1 per-col, 2 per-row
    float* __restrict__ C, long long sC, int ldc,
    int M, int N, int K, int act)                     // act: 0 none, 1 relu, 2 tanh
{
    __builtin_amdgcn_s_wait_tensorcnt(0);
    __shared__ __align__(16) __bf16 smem[2 * BUFSZ];  // double-buffered A+B tiles (80KB)

    const int b = blockIdx.z;
    const __bf16* Ab = A + (long long)b * sA;
    const __bf16* Bb = B + (long long)b * sB;
    float*        Cb = C + (long long)b * sC;
    const int m0 = blockIdx.y * BLKM;
    const int n0 = blockIdx.x * BLKN;
    const int tid  = threadIdx.x;
    const int lane = tid & 31, wave = tid >> 5;
    const int half = lane >> 4, l16 = lane & 15;

    v8f acc[8] = {};

    bool pipe = false;
#ifdef ASYNC_LDS
    pipe = (m0 + BLKM <= M) && (n0 + BLKN <= N) && ((K & (KC - 1)) == 0);
#endif

    if (pipe) {
#ifdef ASYNC_LDS
        const int nch = K / KC;
        stage_async(Ab, lda, Bb, ldb, m0, n0, 0, smem, smem + BLKM * LDSA, tid);
        wait_async_all();
        __syncthreads();
        for (int c = 0; c < nch; ++c) {
            const int cur = c & 1;
            __bf16* curb = smem + cur * BUFSZ;
            if (c + 1 < nch) {   // overlap: copy chunk c+1 while computing chunk c
                __bf16* nxtb = smem + (cur ^ 1) * BUFSZ;
                stage_async(Ab, lda, Bb, ldb, m0, n0, (c + 1) * KC,
                            nxtb, nxtb + BLKM * LDSA, tid);
            }
            tile_compute(curb, curb + BLKM * LDSA, acc, wave, half, l16);
            wait_async_all();
            __syncthreads();
        }
#endif
    } else {
        __bf16* As = smem;
        __bf16* Bs = smem + BLKM * LDSA;
        for (int k0 = 0; k0 < K; k0 += KC) {
            if (k0 + KC < K) {   // global_prefetch_b8 for the next K-chunk
                __builtin_prefetch(Ab + (long long)(k0 + KC), 0, 0);
                __builtin_prefetch(Bb + (long long)(k0 + KC), 0, 0);
            }
            stage_scalar(Ab, lda, Bb, ldb, m0, n0, k0, M, N, K, As, Bs, tid);
            __syncthreads();
            tile_compute(As, Bs, acc, wave, half, l16);
            __syncthreads();
        }
    }

    const int rowb = m0 + wave * 16 + half * 8;       // C/D layout: VGPR r -> M = half*8+r
    #pragma unroll
    for (int nt = 0; nt < 8; ++nt) {
        int col = n0 + nt * 16 + l16;
        if (col >= N) continue;
        #pragma unroll
        for (int r = 0; r < 8; ++r) {
            int row = rowb + r;
            if (row >= M) continue;
            float v = acc[nt][r];
            if (biasMode == 1) v += bias[col];
            else if (biasMode == 2) v += bias[row];
            if (act == 1) v = fmaxf(v, 0.f);
            else if (act == 2) v = tanhf(v);
            Cb[(long long)row * ldc + col] = v;
        }
    }
}

// ---------------- 5) adjacency normalization helpers --------------------------------
__global__ __launch_bounds__(256) void diag_kernel(const float* __restrict__ G,
                                                   float* __restrict__ nrmD)
{
    int b = blockIdx.x, t = threadIdx.x;
    long long d = ((long long)b * NP_ + t);
    nrmD[d] = sqrtf(fmaxf(G[d * NP_ + t], 0.f));
}

__global__ __launch_bounds__(256) void rowsum_kernel(const float* __restrict__ G,
                                                     const float* __restrict__ nrmD,
                                                     float* __restrict__ dinv)
{
    __shared__ float sh[256];
    long long rowid = blockIdx.x;               // b*NP_ + n
    int b = (int)(rowid >> 8);
    int n = (int)(rowid & 255);
    int t = threadIdx.x;
    const float* gr = G + rowid * NP_;
    float nn = nrmD[rowid];
    float nm = nrmD[(long long)b * NP_ + t];
    float v = gr[t] / (nn * nm) + (t == n ? 1.f : 0.f);
    float sum = redSum(sh, 256, v);
    if (t == 0) dinv[rowid] = 1.f / sqrtf(sum);
}

__global__ __launch_bounds__(256) void ahat_kernel(float* __restrict__ G,  // in-place
                                                   const float* __restrict__ nrmD,
                                                   const float* __restrict__ dinv)
{
    long long rowid = blockIdx.x;
    int b = (int)(rowid >> 8);
    int n = (int)(rowid & 255);
    int t = threadIdx.x;
    float* gr = G + rowid * NP_;
    float nn = nrmD[rowid];
    float nm = nrmD[(long long)b * NP_ + t];
    float v = gr[t] / (nn * nm) + (t == n ? 1.f : 0.f);
    gr[t] = dinv[rowid] * dinv[(long long)b * NP_ + t] * v;
}

// ---------------- 6) softmax over n + gating multiply -------------------------------
__global__ __launch_bounds__(256) void softmax_mul_kernel(const float* __restrict__ spre,
                                                          const float* __restrict__ h,
                                                          float* __restrict__ outh)
{
    __shared__ float sh[256];
    long long rowid = blockIdx.x;               // b*HID_ + hh
    int b  = (int)(rowid >> 8);
    int hh = (int)(rowid & 255);
    int t  = threadIdx.x;                       // t = n
    const float* rp = spre + rowid * NP_;
    float v  = rp[t];
    float mx = redMax(sh, 256, v);
    float e  = expf(v - mx);
    float sum = redSum(sh, 256, e);
    float sgate = e / sum;
    long long o = ((long long)b * NP_ + t) * HID_ + hh;
    outh[o] = h[o] * sgate;
}

// ---------------- 7) final fully-connected reduction --------------------------------
__global__ __launch_bounds__(256) void fc_kernel(const float* __restrict__ hfin,
                                                 const float* __restrict__ W,
                                                 const float* __restrict__ bvec,
                                                 float* __restrict__ out)
{
    __shared__ float sh[256];
    int b = blockIdx.x, t = threadIdx.x;
    long long base = (long long)b * (NP_ * HID_);
    float acc = 0.f;
    for (int i = t; i < NP_ * HID_; i += 256) acc += hfin[base + i] * W[i];
    float sum = redSum(sh, 256, acc);
    if (t == 0) out[b] = sum + bvec[0];
}

// ---------------- host-side orchestration -------------------------------------------
extern "C" void kernel_launch(void* const* d_in, const int* in_sizes, int n_in,
                              void* d_out, int out_size, void* d_ws, size_t ws_size,
                              hipStream_t stream)
{
    const float* x      = (const float*)d_in[0];
    const float* gcn_W  = (const float*)d_in[1];
    const float* gcn_b  = (const float*)d_in[2];
    const float* p1_pW  = (const float*)d_in[3];
    const float* p1_pb  = (const float*)d_in[4];
    const float* p1_lW  = (const float*)d_in[5];
    const float* p1_lb  = (const float*)d_in[6];
    const float* p2_pW  = (const float*)d_in[7];
    const float* p2_pb  = (const float*)d_in[8];
    const float* p2_lW  = (const float*)d_in[9];
    const float* p2_lb  = (const float*)d_in[10];
    const float* at_W   = (const float*)d_in[11];
    const float* at_b   = (const float*)d_in[12];
    const float* as_W   = (const float*)d_in[13];
    const float* as_b   = (const float*)d_in[14];
    const float* fc_W   = (const float*)d_in[15];
    const float* fc_b   = (const float*)d_in[16];
    float* out = (float*)d_out;
    (void)in_sizes; (void)n_in; (void)out_size; (void)ws_size;

    char* wsp = (char*)d_ws;
    auto carveF = [&](size_t elems) -> float* {
        float* p = (float*)wsp;
        wsp += (elems * 4 + 255) & ~(size_t)255;
        return p;
    };
    auto carveH = [&](size_t elems) -> __bf16* {
        __bf16* p = (__bf16*)wsp;
        wsp += (elems * 2 + 255) & ~(size_t)255;
        return p;
    };
    // fp32 buffers
    float* featRaw = carveF((size_t)BS_ * NP_ * 20);
    float* feat40  = carveF((size_t)BS_ * NP_ * 40);
    float* G       = carveF((size_t)BS_ * NP_ * NP_);   // becomes Ahat in place
    float* nrmD    = carveF((size_t)BS_ * NP_);
    float* dinv    = carveF((size_t)BS_ * NP_);
    float* M2      = carveF((size_t)BS_ * NP_ * IND_);
    float* bufA    = carveF((size_t)BS_ * NP_ * HID_);
    float* bufB    = carveF((size_t)BS_ * NP_ * HID_);
    float* aBuf    = carveF((size_t)BS_ * HID_ * ATT_);
    float* spre    = carveF((size_t)BS_ * HID_ * NP_);
    // bf16 K-major operand buffers
    __bf16* Fb     = carveH((size_t)BS_ * NP_ * IND_);   // [L][40]
    __bf16* FbT    = carveH((size_t)BS_ * IND_ * NP_);   // [40][L]
    __bf16* AhB    = carveH((size_t)BS_ * NP_ * NP_);
    __bf16* M2b    = carveH((size_t)BS_ * NP_ * IND_);
    __bf16* actN   = carveH((size_t)BS_ * NP_ * HID_);   // natural-layout activations
    __bf16* actT   = carveH((size_t)BS_ * NP_ * HID_);   // transposed activations
    __bf16* aBufB  = carveH((size_t)BS_ * HID_ * ATT_);
    __bf16* gcn_WT = carveH((size_t)HID_ * IND_);
    __bf16* pWT1   = carveH((size_t)NP_ * NP_);
    __bf16* lWT1   = carveH((size_t)HID_ * HID_);
    __bf16* pWT2   = carveH((size_t)NP_ * NP_);
    __bf16* lWT2   = carveH((size_t)HID_ * HID_);
    __bf16* at_WT  = carveH((size_t)ATT_ * NP_);
    __bf16* as_WT  = carveH((size_t)NP_ * ATT_);

    auto pack = [&](const float* src, long long sS, int ldsrc,
                    __bf16* dst, long long sD, int R, int C, int trans, int nb) {
        dim3 grid((unsigned)(((long long)R * C + 255) / 256), (unsigned)nb);
        pack_bf16_kernel<<<grid, 256, 0, stream>>>(src, sS, ldsrc, dst, sD, R, C, trans);
    };
    auto gemm = [&](const __bf16* A, long long sA, int lda,
                    const __bf16* B, long long sB, int ldb,
                    const float* bias, int bm,
                    float* C, long long sC, int ldc,
                    int M, int N, int K, int act) {
        dim3 grid((N + BLKN - 1) / BLKN, (M + BLKM - 1) / BLKM, BS_);
        gemm_bf16_kernel<<<grid, 256, 0, stream>>>(A, sA, lda, B, sB, ldb,
                                                   bias, bm, C, sC, ldc, M, N, K, act);
    };

    // pack static weights (transposed to [out_row][K] K-major form)
    pack(gcn_W, 0, HID_, gcn_WT, 0, HID_, IND_, 1, 1);   // [hid][feat]  = gcn_W[feat][hid]
    pack(p1_pW, 0, NP_,  pWT1,   0, NP_,  NP_,  1, 1);   // [m][n]       = pW[n][m]
    pack(p1_lW, 0, HID_, lWT1,   0, HID_, HID_, 1, 1);   // [hout][hin]  = lW[hin][hout]
    pack(p2_pW, 0, NP_,  pWT2,   0, NP_,  NP_,  1, 1);
    pack(p2_lW, 0, HID_, lWT2,   0, HID_, HID_, 1, 1);
    pack(at_W,  0, ATT_, at_WT,  0, ATT_, NP_,  1, 1);   // [att][n]     = at_W[n][att]
    pack(as_W,  0, NP_,  as_WT,  0, NP_,  ATT_, 1, 1);   // [n][att]     = as_W[att][n]

    // 1-3: features -> 40-d normalized per-batch features
    features_kernel<<<BS_ * NP_, 128, 0, stream>>>(x, featRaw);
    cumsum_kernel<<<(BS_ * 20 + 255) / 256, 256, 0, stream>>>(featRaw, feat40);
    norm_kernel<<<BS_, 256, 0, stream>>>(feat40);
    pack(feat40, (long long)NP_ * IND_, IND_, Fb,  (long long)NP_ * IND_, NP_, IND_, 0, BS_);
    pack(feat40, (long long)NP_ * IND_, IND_, FbT, (long long)IND_ * NP_, IND_, NP_, 1, BS_);

    // 4: G = F * F^T  (M=N=256, K=40)
    gemm(Fb, (long long)NP_ * IND_, IND_, Fb, (long long)NP_ * IND_, IND_,
         nullptr, 0, G, (long long)NP_ * NP_, NP_, NP_, NP_, IND_, 0);

    // 5: cosine + self-loop + symmetric degree normalization (in place on G)
    diag_kernel<<<BS_, 256, 0, stream>>>(G, nrmD);
    rowsum_kernel<<<BS_ * NP_, 256, 0, stream>>>(G, nrmD, dinv);
    ahat_kernel<<<BS_ * NP_, 256, 0, stream>>>(G, nrmD, dinv);
    pack(G, (long long)NP_ * NP_, NP_, AhB, (long long)NP_ * NP_, NP_, NP_, 0, BS_);

    // 6: M2 = Ahat * F  (M=256, N=40, K=256)
    gemm(AhB, (long long)NP_ * NP_, NP_, FbT, (long long)IND_ * NP_, NP_,
         nullptr, 0, M2, (long long)NP_ * IND_, IND_, NP_, IND_, NP_, 0);
    pack(M2, (long long)NP_ * IND_, IND_, M2b, (long long)NP_ * IND_, NP_, IND_, 0, BS_);

    // 7: h = relu(M2 @ gcn_W + gcn_b)  -> bufA
    gemm(M2b, (long long)NP_ * IND_, IND_, gcn_WT, 0, IND_,
         gcn_b, 1, bufA, (long long)NP_ * HID_, HID_, NP_, HID_, IND_, 1);

    // 8: proj1
    pack(bufA, (long long)NP_ * HID_, HID_, actT, (long long)NP_ * HID_, HID_, NP_, 1, BS_);
    gemm(pWT1, 0, NP_, actT, (long long)NP_ * HID_, NP_,
         p1_pb, 2, bufB, (long long)NP_ * HID_, HID_, NP_, HID_, NP_, 0);
    pack(bufB, (long long)NP_ * HID_, HID_, actN, (long long)NP_ * HID_, NP_, HID_, 0, BS_);
    gemm(actN, (long long)NP_ * HID_, HID_, lWT1, 0, HID_,
         p1_lb, 1, bufA, (long long)NP_ * HID_, HID_, NP_, HID_, HID_, 1);

    // 9: proj2
    pack(bufA, (long long)NP_ * HID_, HID_, actT, (long long)NP_ * HID_, HID_, NP_, 1, BS_);
    gemm(pWT2, 0, NP_, actT, (long long)NP_ * HID_, NP_,
         p2_pb, 2, bufB, (long long)NP_ * HID_, HID_, NP_, HID_, NP_, 0);
    pack(bufB, (long long)NP_ * HID_, HID_, actN, (long long)NP_ * HID_, NP_, HID_, 0, BS_);
    gemm(actN, (long long)NP_ * HID_, HID_, lWT2, 0, HID_,
         p2_lb, 1, bufA, (long long)NP_ * HID_, HID_, NP_, HID_, HID_, 1);

    // 10: a = tanh(h^T @ at_W + at_b)   (M=HID, N=ATT, K=NP)
    pack(bufA, (long long)NP_ * HID_, HID_, actT, (long long)NP_ * HID_, HID_, NP_, 1, BS_);
    gemm(actT, (long long)NP_ * HID_, NP_, at_WT, 0, NP_,
         at_b, 1, aBuf, (long long)HID_ * ATT_, ATT_, HID_, ATT_, NP_, 2);

    // 11: spre = a @ as_W + as_b   (M=HID, N=NP, K=ATT)
    pack(aBuf, (long long)HID_ * ATT_, ATT_, aBufB, (long long)HID_ * ATT_, HID_, ATT_, 0, BS_);
    gemm(aBufB, (long long)HID_ * ATT_, ATT_, as_WT, 0, ATT_,
         as_b, 1, spre, (long long)HID_ * NP_, NP_, HID_, NP_, ATT_, 0);

    // 12: softmax over n, gate h -> bufB
    softmax_mul_kernel<<<BS_ * HID_, 256, 0, stream>>>(spre, bufA, bufB);

    // 13: out = flat(h) @ fc_W + fc_b
    fc_kernel<<<BS_, 256, 0, stream>>>(bufB, fc_W, fc_b, out);
}
